// NZ_OrderAttn_16939351016276
// MI455X (gfx1250) — matrix operand order repaired and checked by reading, not compiled, not run on hardware
//
#include <hip/hip_runtime.h>
#include <hip/hip_bf16.h>

typedef __attribute__((ext_vector_type(16))) _Float16 v16h;
typedef __attribute__((ext_vector_type(8)))  _Float16 v8h;
typedef __attribute__((ext_vector_type(8)))  float    v8f;

#define EPSN 1e-12f

// ---------------------------------------------------------------- converts
__global__ void f32_to_f16_kernel(const float* __restrict__ in,
                                  _Float16* __restrict__ out, long long n) {
  long long i = (long long)blockIdx.x * blockDim.x + threadIdx.x;
  long long stride = (long long)gridDim.x * blockDim.x;
  for (; i < n; i += stride) out[i] = (_Float16)in[i];
}

// ------------------------------------------------------------- row L2 norms
__global__ void row_norm_kernel(const _Float16* __restrict__ X,
                                float* __restrict__ out,
                                int rowsPerBatch, int batchRowStride, int m) {
  int r = blockIdx.x;
  int b = r / rowsPerBatch, c = r % rowsPerBatch;
  const _Float16* p = X + ((long long)b * batchRowStride + c) * (long long)m;
  float s = 0.f;
  for (int i = threadIdx.x * 8; i < m; i += blockDim.x * 8) {
    v8h v = *(const v8h*)&p[i];
    #pragma unroll
    for (int j = 0; j < 8; ++j) { float f = (float)v[j]; s += f * f; }
  }
  __shared__ float red[256];
  red[threadIdx.x] = s;
  __syncthreads();
  for (int off = 128; off > 0; off >>= 1) {
    if (threadIdx.x < off) red[threadIdx.x] += red[threadIdx.x + off];
    __syncthreads();
  }
  if (threadIdx.x == 0) out[r] = sqrtf(red[0]);
}

// ------------------------------------------------- scaled softmax over 64
__global__ void softmax_kernel(const float* __restrict__ gram,
                               const float* __restrict__ nq,
                               const float* __restrict__ nk,
                               const float* __restrict__ temp,
                               _Float16* __restrict__ attn) {
  int row = blockIdx.x;           // bh*64 + i
  int bh = row >> 6, i = row & 63;
  int b = bh >> 3, h = bh & 7;
  int j = threadIdx.x;            // 0..63
  float qn = fmaxf(nq[b * 512 + h * 64 + i], EPSN);
  float kn = fmaxf(nk[b * 512 + h * 64 + j], EPSN);
  float logit = gram[((long long)bh * 64 + i) * 64 + j] / (qn * kn) * temp[h];
  __shared__ float sh[64];
  sh[j] = logit;
  __syncthreads();
  float mx = sh[0];
  for (int t = 1; t < 64; ++t) mx = fmaxf(mx, sh[t]);
  __syncthreads();
  float e = expf(logit - mx);
  sh[j] = e;
  __syncthreads();
  float sum = 0.f;
  for (int t = 0; t < 64; ++t) sum += sh[t];
  attn[((long long)bh * 64 + i) * 64 + j] = (_Float16)(e / sum);
}

// ----------------------------------------------------- generic WMMA GEMM
// D[M x N] = A[M x K] * B[K x N] per batch (grid.z), f32 accumulate.
// Block tile 64x64, K-step 64, 256 threads = 8 waves (4 m-slices x 2 n-slices,
// each wave: 2 accumulators, 4 v_wmma_f32_16x16x32_f16 per K-step).
// Double-buffered LDS + register pipelining: next tile is fetched to VGPRs
// while WMMAs consume the current tile; one barrier per K-step.
// BMODE: 0 plain B[k*ldb+n] | 2 transposed B[n*ldb+k] | 1 conv3x3 gather from
//        zero-padded [512][66][66] activations (K index = c*9 + (dr*3+dc);
//        a 64-wide N tile is one spatial row -> contiguous window loads)
// CMODE: 0 f16 plain | 1 f16 into padded [512][66][66] | 2 f32 plain
template <int BMODE, int CMODE>
__global__ __launch_bounds__(256)
void wmma_gemm_kernel(const _Float16* __restrict__ A,
                      const _Float16* __restrict__ B,
                      void* __restrict__ C,
                      int K, int lda, int ldb, int ldc, int inner,
                      long long soA, long long siA,
                      long long soB, long long siB,
                      long long soC, long long siC) {
  __shared__ alignas(32) _Float16 lA[2][64 * 64];
  __shared__ alignas(32) _Float16 lB[2][64 * 64];

  const int tid  = threadIdx.x;
  const int lane = tid & 31;
  const int wid  = tid >> 5;
  const int wm   = wid & 3;   // 16-row slice
  const int wn   = wid >> 2;  // 32-col slice

  const long long bz    = blockIdx.z;
  const long long outer = bz / inner, innerIdx = bz % inner;
  const _Float16* Ab = A + outer * soA + innerIdx * siA;
  const _Float16* Bb = B + outer * soB + innerIdx * siB;
  const long long cOff = outer * soC + innerIdx * siC;

  const int mBase = blockIdx.y * 64;
  const int nBase = blockIdx.x * 64;

  // staging coordinates: each thread moves 2 x v8h per tile (rows r, r+32)
  const int sR = tid >> 3;          // 0..31
  const int sC = (tid & 7) * 8;     // 0..56
  const int tN  = tid >> 2;         // 0..63   (transposed-B mapping)
  const int tKs = (tid & 3) * 16;   // 0..48

  v8h ra0, ra1, rb0, rb1;

  auto loadStage = [&](int kq) {
    ra0 = *(const v8h*)&Ab[(long long)(mBase + sR) * lda + kq + sC];
    ra1 = *(const v8h*)&Ab[(long long)(mBase + sR + 32) * lda + kq + sC];
    if constexpr (BMODE == 0) {
      rb0 = *(const v8h*)&Bb[(long long)(kq + sR) * ldb + nBase + sC];
      rb1 = *(const v8h*)&Bb[(long long)(kq + sR + 32) * ldb + nBase + sC];
    } else if constexpr (BMODE == 2) {
      const _Float16* src = &Bb[(long long)(nBase + tN) * ldb + kq + tKs];
      rb0 = *(const v8h*)src;
      rb1 = *(const v8h*)(src + 8);
    } else {
      int rr = nBase >> 6;          // 64-wide N tile = one spatial row
      #pragma unroll
      for (int half = 0; half < 2; ++half) {
        int kg = kq + sR + half * 32;
        int cch = kg / 9, t = kg - cch * 9;
        int dr = t / 3, dc = t - dr * 3;
        const _Float16* src =
            &Bb[(long long)cch * 4356 + (rr + dr) * 66 + dc + sC];
        v8h tv;                     // 8 adjacent u16 loads (2B-aligned only)
        #pragma unroll
        for (int i = 0; i < 8; ++i) tv[i] = src[i];
        if (half == 0) rb0 = tv; else rb1 = tv;
      }
    }
  };

  auto storeStage = [&](int ph) {
    _Float16* pa = &lA[ph][0];
    _Float16* pb = &lB[ph][0];
    *(v8h*)&pa[sR * 64 + sC] = ra0;
    *(v8h*)&pa[(sR + 32) * 64 + sC] = ra1;
    if constexpr (BMODE == 2) {     // scatter into [k][n] layout
      #pragma unroll
      for (int i = 0; i < 8; ++i) pb[(tKs + i) * 64 + tN] = rb0[i];
      #pragma unroll
      for (int i = 0; i < 8; ++i) pb[(tKs + 8 + i) * 64 + tN] = rb1[i];
    } else {
      *(v8h*)&pb[sR * 64 + sC] = rb0;
      *(v8h*)&pb[(sR + 32) * 64 + sC] = rb1;
    }
  };

  v8f acc[2];
  #pragma unroll
  for (int t = 0; t < 2; ++t)
    #pragma unroll
    for (int i = 0; i < 8; ++i) acc[t][i] = 0.f;

  loadStage(0);
  storeStage(0);
  __syncthreads();

  int ph = 0;
  for (int k0 = 0; k0 < K; k0 += 64) {
    const int kn = k0 + 64;
    if (kn < K) loadStage(kn);      // overlaps with the WMMAs below

    #pragma unroll
    for (int kk = 0; kk < 2; ++kk) {
      // A fragment (ISA layout): lanes 0-15 M=lane kb=0, lanes 16-31 kb=8
      union { v16h v; _Float16 h[16]; } af;
      {
        int mr = lane & 15;
        int kb = (lane >> 4) * 8;
        const _Float16* arow = &lA[ph][(wm * 16 + mr) * 64 + kk * 32];
        #pragma unroll
        for (int v = 0; v < 8; ++v) {
          int K0 = kb + 2 * v + (v >= 4 ? 8 : 0);
          af.h[2 * v]     = arow[K0];
          af.h[2 * v + 1] = arow[K0 + 1];
        }
      }
      // B fragments: N = lane&15, K in same (table) order as A
      union { v16h v; _Float16 h[16]; } bf0, bf1;
      {
        int nc = lane & 15;
        int kb = (lane >> 4) * 8;
        const _Float16* bcol = &lB[ph][(kk * 32) * 64 + wn * 32 + nc];
        #pragma unroll
        for (int s = 0; s < 16; ++s) {
          int K0 = kb + s + (s >= 8 ? 8 : 0);
          bf0.h[s] = bcol[K0 * 64];
          bf1.h[s] = bcol[K0 * 64 + 16];
        }
      }
      v16h b0 = bf0.v;
      v16h b1 = bf1.v;

      acc[0] = __builtin_amdgcn_wmma_f32_16x16x32_f16(
          false, af.v, false, b0, (short)0, acc[0], false, false);
      acc[1] = __builtin_amdgcn_wmma_f32_16x16x32_f16(
          false, af.v, false, b1, (short)0, acc[1], false, false);
    }

    if (kn >= K) break;
    storeStage(ph ^ 1);
    __syncthreads();
    ph ^= 1;
  }

  // ---- store: VGPR r, lanes 0-15 -> M=r, lanes 16-31 -> M=r+8; N = lane&15
  const int rHi = (lane >> 4) * 8;
  const int cIn = lane & 15;
  #pragma unroll
  for (int t = 0; t < 2; ++t) {
    #pragma unroll
    for (int r = 0; r < 8; ++r) {
      int gr = mBase + wm * 16 + r + rHi;
      int gc = nBase + wn * 32 + t * 16 + cIn;
      float vv = acc[t][r];
      if constexpr (CMODE == 0) {
        ((_Float16*)C)[cOff + (long long)gr * ldc + gc] = (_Float16)vv;
      } else if constexpr (CMODE == 2) {
        ((float*)C)[cOff + (long long)gr * ldc + gc] = vv;
      } else { // padded [512][66][66] store (conv input staging)
        int rr = gc >> 6, cl = gc & 63;
        ((_Float16*)C)[cOff + (long long)gr * 4356 + (rr + 1) * 66 + (cl + 1)]
            = (_Float16)vv;
      }
    }
  }
}

// ---------------------------------------------------------------- launcher
extern "C" void kernel_launch(void* const* d_in, const int* in_sizes, int n_in,
                              void* d_out, int out_size, void* d_ws, size_t ws_size,
                              hipStream_t stream) {
  (void)in_sizes; (void)n_in; (void)out_size; (void)ws_size;
  const float* x     = (const float*)d_in[0];   // [4][512][4096]
  const float* y     = (const float*)d_in[1];   // [4][512][64][64]
  const float* temp  = (const float*)d_in[2];   // [8]
  const float* w_kv  = (const float*)d_in[3];   // [1024][512]
  const float* w_q   = (const float*)d_in[4];   // [512][512]
  const float* w_dw  = (const float*)d_in[5];   // [512][512][3][3] == [512][4608]
  const float* w_out = (const float*)d_in[6];   // [512][512]
  float* out = (float*)d_out;                   // [4][512][4096]

  const long long BN = 4, DIM = 512, M = 4096;
  const long long PAD = 4356; // 66*66

  char* p = (char*)d_ws;
  auto carve = [&](size_t bytes) {
    char* r = p; p += (bytes + 255) & ~(size_t)255; return r;
  };
  _Float16* x_h    = (_Float16*)carve(BN * DIM * M * 2);
  _Float16* y_h    = (_Float16*)carve(BN * DIM * M * 2);
  _Float16* wkv_h  = (_Float16*)carve(1024 * 512 * 2);
  _Float16* wq_h   = (_Float16*)carve(512 * 512 * 2);
  _Float16* wdw_h  = (_Float16*)carve(512 * 4608 * 2);
  _Float16* wout_h = (_Float16*)carve(512 * 512 * 2);
  _Float16* kv_h   = (_Float16*)carve(BN * 1024 * M * 2);   // k rows 0..511, v rows 512..1023
  _Float16* q1p_h  = (_Float16*)carve(BN * DIM * PAD * 2);  // zero-padded conv input
  _Float16* qc_h   = (_Float16*)carve(BN * DIM * M * 2);    // conv output (the "q")
  _Float16* aout_h = (_Float16*)carve(BN * DIM * M * 2);    // attn @ v
  float*    gram   = (float*)   carve(32 * 64 * 64 * 4);
  float*    nq     = (float*)   carve(BN * DIM * 4);
  float*    nk     = (float*)   carve(BN * DIM * 4);
  _Float16* attn_h = (_Float16*)carve(32 * 64 * 64 * 2);

  // 1) f32 -> f16 conversions
  f32_to_f16_kernel<<<2048, 256, 0, stream>>>(x, x_h, BN * DIM * M);
  f32_to_f16_kernel<<<2048, 256, 0, stream>>>(y, y_h, BN * DIM * M);
  f32_to_f16_kernel<<<512, 256, 0, stream>>>(w_kv, wkv_h, 1024 * 512);
  f32_to_f16_kernel<<<256, 256, 0, stream>>>(w_q, wq_h, 512 * 512);
  f32_to_f16_kernel<<<1024, 256, 0, stream>>>(w_dw, wdw_h, 512 * 4608);
  f32_to_f16_kernel<<<256, 256, 0, stream>>>(w_out, wout_h, 512 * 512);
  hipMemsetAsync(q1p_h, 0, (size_t)(BN * DIM * PAD * 2), stream);

  // 2) kv = w_kv @ x : [1024x512]x[512x4096] per batch -> kv_h
  wmma_gemm_kernel<0, 0><<<dim3(64, 16, 4), 256, 0, stream>>>(
      wkv_h, x_h, kv_h, 512, 512, 4096, 4096, 1,
      0, 0, 512 * 4096LL, 0, 1024 * 4096LL, 0);

  // 3) q1 = w_q @ y : [512x512]x[512x4096], store into padded [512][66][66]
  //    (the reference's linear interp is identity since h*w == m)
  wmma_gemm_kernel<0, 1><<<dim3(64, 8, 4), 256, 0, stream>>>(
      wq_h, y_h, q1p_h, 512, 512, 4096, 0, 1,
      0, 0, 512 * 4096LL, 0, DIM * PAD, 0);

  // 4) q = conv3x3(q1) as implicit GEMM: [512x4608]x[4608x4096] per batch
  wmma_gemm_kernel<1, 0><<<dim3(64, 8, 4), 256, 0, stream>>>(
      wdw_h, q1p_h, qc_h, 4608, 4608, 0, 4096, 1,
      0, 0, DIM * PAD, 0, 512 * 4096LL, 0);

  // 5) row L2 norms of q and k (f16-consistent with the Gram inputs)
  row_norm_kernel<<<2048, 256, 0, stream>>>(qc_h, nq, 512, 512, 4096);
  row_norm_kernel<<<2048, 256, 0, stream>>>(kv_h, nk, 512, 1024, 4096);

  // 6) Gram: S = Q * K^T (64x64, K=4096) per (b,head); inner batch = 8 heads
  wmma_gemm_kernel<2, 2><<<dim3(1, 1, 32), 256, 0, stream>>>(
      qc_h, kv_h, gram, 4096, 4096, 4096, 64, 8,
      512 * 4096LL, 64 * 4096LL,      // A: qc rows (b*512 + h*64)
      1024 * 4096LL, 64 * 4096LL,     // B: k rows (b*1024 + h*64), transposed
      8 * 4096LL, 4096LL);            // C: gram[bh][64][64]

  // 7) scaled softmax -> attn (f16)
  softmax_kernel<<<2048, 64, 0, stream>>>(gram, nq, nk, temp, attn_h);

  // 8) out = attn @ v : [64x64]x[64x4096] per (b,head)
  wmma_gemm_kernel<0, 0><<<dim3(64, 1, 32), 256, 0, stream>>>(
      attn_h, kv_h + 512 * 4096LL, aout_h, 64, 64, 4096, 4096, 8,
      8 * 4096LL, 4096LL,             // A: attn[bh]
      1024 * 4096LL, 64 * 4096LL,     // B: v rows (b*1024 + 512 + h*64)
      512 * 4096LL, 64 * 4096LL);     // C: aout rows (b*512 + h*64)

  // 9) final projection: w_out @ out -> d_out (f32)
  wmma_gemm_kernel<0, 2><<<dim3(64, 8, 4), 256, 0, stream>>>(
      wout_h, aout_h, out, 512, 512, 4096, 4096, 1,
      0, 0, 512 * 4096LL, 0, 512 * 4096LL, 0);
}
